// NewModel_67293547593937
// MI455X (gfx1250) — matrix operand order repaired and checked by reading, not compiled, not run on hardware
//
#include <hip/hip_runtime.h>
#include <hip/hip_bf16.h>

#define HDIM 128
#define NNODE 65536
#define NEDGE 250000
#define NGRAPH 256
#define H3 384

typedef __attribute__((ext_vector_type(16))) __bf16 v16bf;
typedef __attribute__((ext_vector_type(8)))  float  v8f;

// ---- CDNA5 async memory->LDS helpers (ASYNCcnt tracked) ----
__device__ __forceinline__ void async_ld_b128(unsigned lds_off, const void* g)
{
    asm volatile("global_load_async_to_lds_b128 %0, %1, off"
                 :: "v"(lds_off), "v"(g) : "memory");
}
__device__ __forceinline__ void wait_async0()
{
    asm volatile("s_wait_asynccnt 0x0" ::: "memory");
}

// ---------------- bf16 WMMA GEMM: Y[M,128] = op(X[M,128] @ W + bias [+Y]) ----------------
// X: bf16 row-major [M][128]. Wt: bf16 TRANSPOSED [n][k] (Wt[j*128+k] = W[k][j]).
__global__ __launch_bounds__(256)
void gemm128_bias(const __bf16* __restrict__ X, const __bf16* __restrict__ Wt,
                  const float* __restrict__ bias, float* __restrict__ Y,
                  int accumulate, int do_relu)
{
    __shared__ __bf16 Ash[128][136];   // [m][k], row stride 272B (pad keeps 16B chunks in-row)
    __shared__ __bf16 Bsh[128][136];   // [n][k]

    const int tid  = threadIdx.x;
    const int row0 = blockIdx.x * 128;

    const unsigned a_base = (unsigned)(size_t)(&Ash[0][0]);
    const unsigned b_base = (unsigned)(size_t)(&Bsh[0][0]);

    // 128 rows x 256B each = 2048 x b128 chunks per matrix; 8 per thread
    #pragma unroll
    for (int i = 0; i < 8; ++i) {
        int chunk = tid + i * 256;          // 0..2047
        int m   = chunk >> 4;               // 0..127
        int k16 = chunk & 15;               // 16B chunk within row
        async_ld_b128(a_base + (unsigned)(m * 272 + k16 * 16),
                      X + (size_t)(row0 + m) * HDIM + k16 * 8);
        async_ld_b128(b_base + (unsigned)(m * 272 + k16 * 16),
                      Wt + (size_t)m * HDIM + k16 * 8);
    }
    wait_async0();
    __syncthreads();

    const int wave  = tid >> 5;
    const int lane  = tid & 31;
    const int lrow  = lane & 15;
    const int hf    = lane >> 4;         // lane half (0/1)
    const int mbase = wave * 16;

    v8f acc[8] = {};

    #pragma unroll
    for (int kk = 0; kk < 128; kk += 32) {
        // A fragment: contiguous k = kk + 16*half + s (same order as B)
        v16bf afrag;
        #pragma unroll
        for (int p = 0; p < 8; ++p) {
            int k0 = kk + hf * 16 + 2 * p;
            afrag[2 * p]     = Ash[mbase + lrow][k0];
            afrag[2 * p + 1] = Ash[mbase + lrow][k0 + 1];
        }
        #pragma unroll
        for (int n = 0; n < 8; ++n) {
            // B fragment: lane = column N, half selects K 0-15 / 16-31
            v16bf bfrag;
            #pragma unroll
            for (int v = 0; v < 8; ++v) {
                int k0 = kk + hf * 16 + 2 * v;
                bfrag[2 * v]     = Bsh[n * 16 + lrow][k0];
                bfrag[2 * v + 1] = Bsh[n * 16 + lrow][k0 + 1];
            }
            acc[n] = __builtin_amdgcn_wmma_f32_16x16x32_bf16(
                false, afrag, false, bfrag, (short)0, acc[n], false, false);
        }
    }

    // D layout: VGPR r -> row mbase + r + 8*half, col = n*16 + lrow
    #pragma unroll
    for (int n = 0; n < 8; ++n) {
        int col = n * 16 + lrow;
        float bvv = bias ? bias[col] : 0.0f;
        #pragma unroll
        for (int r = 0; r < 8; ++r) {
            int m = mbase + r + 8 * hf;
            size_t o = (size_t)(row0 + m) * HDIM + col;
            float y = acc[n][r] + bvv;
            if (accumulate) y += Y[o];
            if (do_relu) y = fmaxf(y, 0.0f);
            Y[o] = y;
        }
    }
}

// ---------------- edge kernel: one wave per edge, 4 feats/lane ----------------
__global__ __launch_bounds__(256)
void edge_gate_scatter(const int* __restrict__ ei, const float* __restrict__ ea,
                       const float* __restrict__ We, const float* __restrict__ K,
                       const float* __restrict__ Q, const float* __restrict__ V,
                       float* __restrict__ OUT, int E)
{
    int widx = (blockIdx.x * (blockDim.x >> 5)) + (threadIdx.x >> 5);
    if (widx >= E) return;
    int lane = threadIdx.x & 31;
    int s = ei[widx];
    int d = ei[E + widx];
    float a = ea[widx];
    int h0 = lane * 4;

    float4 kv = *(const float4*)(K + (size_t)d * HDIM + h0);
    float4 qv = *(const float4*)(Q + (size_t)s * HDIM + h0);
    float4 vv = *(const float4*)(V + (size_t)s * HDIM + h0);
    float4 wv = *(const float4*)(We + h0);

    float g0 = 1.0f / (1.0f + __expf(-(kv.x + qv.x + a * wv.x)));
    float g1 = 1.0f / (1.0f + __expf(-(kv.y + qv.y + a * wv.y)));
    float g2 = 1.0f / (1.0f + __expf(-(kv.z + qv.z + a * wv.z)));
    float g3 = 1.0f / (1.0f + __expf(-(kv.w + qv.w + a * wv.w)));

    float* o = OUT + (size_t)d * HDIM + h0;
    atomicAdd(o + 0, g0 * vv.x);
    atomicAdd(o + 1, g1 * vv.y);
    atomicAdd(o + 2, g2 * vv.z);
    atomicAdd(o + 3, g3 * vv.w);
}

// ---------------- conversions / elementwise ----------------
__global__ void f32_to_bf16_kernel(const float* __restrict__ src, __bf16* __restrict__ dst, size_t n)
{
    size_t i = (size_t)blockIdx.x * blockDim.x + threadIdx.x;
    size_t stride = (size_t)gridDim.x * blockDim.x;
    for (; i < n; i += stride) dst[i] = (__bf16)src[i];
}

__global__ void relu_to_bf16_kernel(const float* __restrict__ src, __bf16* __restrict__ dst, size_t n)
{
    size_t i = (size_t)blockIdx.x * blockDim.x + threadIdx.x;
    size_t stride = (size_t)gridDim.x * blockDim.x;
    for (; i < n; i += stride) dst[i] = (__bf16)fmaxf(src[i], 0.0f);
}

// transpose+convert 128x128 weight matrices: Wt[mat][j][k] = (bf16)W[mat][k][j]
__global__ void conv_w_kernel(const float* __restrict__ W, __bf16* __restrict__ Wt, int total)
{
    int i = blockIdx.x * blockDim.x + threadIdx.x;
    if (i >= total) return;
    int mat = i >> 14;
    int rem = i & 16383;
    int k = rem >> 7;
    int j = rem & 127;
    Wt[((size_t)mat << 14) + j * HDIM + k] = (__bf16)W[i];
}

__global__ void zero_kernel(float* __restrict__ p, size_t n)
{
    size_t i = (size_t)blockIdx.x * blockDim.x + threadIdx.x;
    size_t stride = (size_t)gridDim.x * blockDim.x;
    for (; i < n; i += stride) p[i] = 0.0f;
}

// ---------------- mean pool (atomic accumulate, relu always applied on read) ----------------
__global__ __launch_bounds__(256)
void pool_accum(const float* __restrict__ h, const int* __restrict__ batch,
                float* __restrict__ sums, float* __restrict__ counts, int N)
{
    int warp = (blockIdx.x * (blockDim.x >> 5)) + (threadIdx.x >> 5);
    if (warp >= N) return;
    int lane = threadIdx.x & 31;
    int g = batch[warp];
    float4 hv = *(const float4*)(h + (size_t)warp * HDIM + lane * 4);
    float* s = sums + (size_t)g * HDIM + lane * 4;
    atomicAdd(s + 0, fmaxf(hv.x, 0.0f));   // h == relu(o)
    atomicAdd(s + 1, fmaxf(hv.y, 0.0f));
    atomicAdd(s + 2, fmaxf(hv.z, 0.0f));
    atomicAdd(s + 3, fmaxf(hv.w, 0.0f));
    if (lane == 0) atomicAdd(&counts[g], 1.0f);
}

__global__ void pool_finalize(const float* __restrict__ sc, const float* __restrict__ cc,
                              const float* __restrict__ sx, const float* __restrict__ cx,
                              const float* __restrict__ sb, const float* __restrict__ cb,
                              float* __restrict__ gbuf)
{
    int g = blockIdx.x;
    int t = threadIdx.x;   // 128
    gbuf[(size_t)g * H3 + t]       = sc[g * HDIM + t] / fmaxf(cc[g], 1.0f);
    gbuf[(size_t)g * H3 + 128 + t] = sx[g * HDIM + t] / fmaxf(cx[g], 1.0f);
    gbuf[(size_t)g * H3 + 256 + t] = sb[g * HDIM + t] / fmaxf(cb[g], 1.0f);
}

// ---------------- MLP head ----------------
__global__ void mlp_layer(const float* __restrict__ in, const float* __restrict__ Wm,
                          const float* __restrict__ bm, float* __restrict__ out, int do_relu)
{
    int j = blockIdx.x * blockDim.x + threadIdx.x;  // 0..383
    int g = blockIdx.y;
    if (j >= H3) return;
    float s = bm[j];
    for (int k = 0; k < H3; ++k) s += in[(size_t)g * H3 + k] * Wm[(size_t)k * H3 + j];
    if (do_relu) s = fmaxf(s, 0.0f);
    out[(size_t)g * H3 + j] = s;
}

__global__ void mlp_out_kernel(const float* __restrict__ in, const float* __restrict__ Wo,
                               const float* __restrict__ bo, float* __restrict__ out)
{
    int g = blockIdx.x * blockDim.x + threadIdx.x;
    if (g >= NGRAPH) return;
    float s = bo[0];
    for (int k = 0; k < H3; ++k) s += in[(size_t)g * H3 + k] * Wo[k];
    out[g] = s;
}

// ---------------- host orchestration ----------------
extern "C" void kernel_launch(void* const* d_in, const int* in_sizes, int n_in,
                              void* d_out, int out_size, void* d_ws, size_t ws_size,
                              hipStream_t stream)
{
    (void)in_sizes; (void)n_in; (void)out_size; (void)ws_size;

    const float* x_x  = (const float*)d_in[0];
    const float* x_b  = (const float*)d_in[1];
    const float* x_c  = (const float*)d_in[2];
    const float* ea_arr[4] = {(const float*)d_in[3], (const float*)d_in[4],
                              (const float*)d_in[5], (const float*)d_in[6]};
    const float* Wk = (const float*)d_in[7];
    const float* bk = (const float*)d_in[8];
    const float* Wq = (const float*)d_in[9];
    const float* bq = (const float*)d_in[10];
    const float* Wv = (const float*)d_in[11];
    const float* bv = (const float*)d_in[12];
    const float* We = (const float*)d_in[13];
    const float* Ws = (const float*)d_in[14];
    const float* cb = (const float*)d_in[15];
    const float* W1 = (const float*)d_in[16];
    const float* b1 = (const float*)d_in[17];
    const float* W2 = (const float*)d_in[18];
    const float* b2 = (const float*)d_in[19];
    const float* W3 = (const float*)d_in[20];
    const float* b3 = (const float*)d_in[21];
    const float* Wo = (const float*)d_in[22];
    const float* bo = (const float*)d_in[23];
    const int* ei_arr[4] = {(const int*)d_in[24], (const int*)d_in[25],
                            (const int*)d_in[26], (const int*)d_in[27]};
    const int* batch_x = (const int*)d_in[28];
    const int* batch_b = (const int*)d_in[29];
    const int* batch_c = (const int*)d_in[30];

    const size_t NH = (size_t)NNODE * HDIM;       // 8388608
    const int    HH = HDIM * HDIM;                // 16384
    const int    NMAT = 12;                       // L*R matrices per weight tensor

    float* ws = (float*)d_ws;
    float* ox = ws + 0 * NH;
    float* ob = ws + 1 * NH;
    float* oc = ws + 2 * NH;
    float* kb = ws + 3 * NH;
    float* qb = ws + 4 * NH;
    float* vb = ws + 5 * NH;

    float* small  = ws + 6 * NH;
    float* sums_c = small;                        // G*128
    float* sums_x = sums_c + NGRAPH * HDIM;
    float* sums_b = sums_x + NGRAPH * HDIM;
    float* cnt_c  = sums_b + NGRAPH * HDIM;       // counts follow contiguously
    float* cnt_x  = cnt_c + NGRAPH;
    float* cnt_b  = cnt_x + NGRAPH;
    float* gA     = cnt_b + NGRAPH;               // G*384
    float* gB     = gA + (size_t)NGRAPH * H3;
    float* small_end = gB + (size_t)NGRAPH * H3;  // 295,680 floats total (16B multiple)

    __bf16* bws = (__bf16*)small_end;
    __bf16* hxb = bws + 0 * NH;
    __bf16* hbb = bws + 1 * NH;
    __bf16* hcb = bws + 2 * NH;
    __bf16* Wkt = bws + 3 * NH;                   // 12 * 16384 bf16 each
    __bf16* Wqt = Wkt + (size_t)NMAT * HH;
    __bf16* Wvt = Wqt + (size_t)NMAT * HH;
    __bf16* Wst = Wvt + (size_t)NMAT * HH;

    // ---- precompute: bf16 features, bf16 transposed weights ----
    f32_to_bf16_kernel<<<8192, 256, 0, stream>>>(x_x, hxb, NH);
    f32_to_bf16_kernel<<<8192, 256, 0, stream>>>(x_b, hbb, NH);
    f32_to_bf16_kernel<<<8192, 256, 0, stream>>>(x_c, hcb, NH);
    const int wtotal = NMAT * HH;                 // 196608
    conv_w_kernel<<<(wtotal + 255) / 256, 256, 0, stream>>>(Wk, Wkt, wtotal);
    conv_w_kernel<<<(wtotal + 255) / 256, 256, 0, stream>>>(Wq, Wqt, wtotal);
    conv_w_kernel<<<(wtotal + 255) / 256, 256, 0, stream>>>(Wv, Wvt, wtotal);
    conv_w_kernel<<<(wtotal + 255) / 256, 256, 0, stream>>>(Ws, Wst, wtotal);

    const int gemm_blocks = NNODE / 128;          // 512
    const int edge_blocks = (NEDGE + 7) / 8;

    for (int l = 0; l < 3; ++l) {
        auto woff = [&](int r) { return (size_t)(l * 4 + r) * HH; };
        auto boff = [&](int r) { return (size_t)(l * 4 + r) * HDIM; };

        // skip connections + conv bias (dst-side): oc gets both relations into 'c'
        gemm128_bias<<<gemm_blocks, 256, 0, stream>>>(hcb, Wst + woff(0), cb + boff(0), oc, 0, 0);
        gemm128_bias<<<gemm_blocks, 256, 0, stream>>>(hcb, Wst + woff(1), cb + boff(1), oc, 1, 0);
        gemm128_bias<<<gemm_blocks, 256, 0, stream>>>(hxb, Wst + woff(2), cb + boff(2), ox, 0, 0);
        gemm128_bias<<<gemm_blocks, 256, 0, stream>>>(hbb, Wst + woff(3), cb + boff(3), ob, 0, 0);

        const __bf16* srcs[4] = {hxb, hbb, hcb, hcb};
        const __bf16* dsts[4] = {hcb, hcb, hxb, hbb};
        float*        outs[4] = {oc, oc, ox, ob};

        for (int r = 0; r < 4; ++r) {
            gemm128_bias<<<gemm_blocks, 256, 0, stream>>>(dsts[r], Wkt + woff(r), bk + boff(r), kb, 0, 0);
            gemm128_bias<<<gemm_blocks, 256, 0, stream>>>(srcs[r], Wqt + woff(r), bq + boff(r), qb, 0, 0);
            gemm128_bias<<<gemm_blocks, 256, 0, stream>>>(srcs[r], Wvt + woff(r), bv + boff(r), vb, 0, 0);
            edge_gate_scatter<<<edge_blocks, 256, 0, stream>>>(
                ei_arr[r], ea_arr[r], We + boff(r), kb, qb, vb, outs[r], NEDGE);
        }

        if (l < 2) {  // next layer consumes bf16 h = relu(o)
            relu_to_bf16_kernel<<<8192, 256, 0, stream>>>(oc, hcb, NH);
            relu_to_bf16_kernel<<<8192, 256, 0, stream>>>(ox, hxb, NH);
            relu_to_bf16_kernel<<<8192, 256, 0, stream>>>(ob, hbb, NH);
        }
    }

    // ---- pooling: read final-layer o buffers, relu applied inside accumulate ----
    size_t zn = (size_t)3 * NGRAPH * HDIM + 3 * NGRAPH;
    zero_kernel<<<256, 256, 0, stream>>>(sums_c, zn);

    const int pool_blocks = NNODE / 8;
    pool_accum<<<pool_blocks, 256, 0, stream>>>(oc, batch_c, sums_c, cnt_c, NNODE);
    pool_accum<<<pool_blocks, 256, 0, stream>>>(ox, batch_x, sums_x, cnt_x, NNODE);
    pool_accum<<<pool_blocks, 256, 0, stream>>>(ob, batch_b, sums_b, cnt_b, NNODE);
    pool_finalize<<<NGRAPH, 128, 0, stream>>>(sums_c, cnt_c, sums_x, cnt_x, sums_b, cnt_b, gA);

    dim3 mlpGrid(3, NGRAPH);
    mlp_layer<<<mlpGrid, 128, 0, stream>>>(gA, W1, b1, gB, 1);
    mlp_layer<<<mlpGrid, 128, 0, stream>>>(gB, W2, b2, gA, 1);
    mlp_layer<<<mlpGrid, 128, 0, stream>>>(gA, W3, b3, gB, 1);
    mlp_out_kernel<<<1, NGRAPH, 0, stream>>>(gB, Wo, bo, (float*)d_out);
}
